// EmotionalEmbeddingSpace_37649683317431
// MI455X (gfx1250) — matrix operand order repaired and checked by reading, not compiled, not run on hardware
//
#include <hip/hip_runtime.h>
#include <hip/hip_bf16.h>

// ---------------------------------------------------------------------------
// Types for CDNA5 WMMA (wave32): V_WMMA_F32_16X16X32_BF16
// ---------------------------------------------------------------------------
typedef __bf16 bf16;
typedef __attribute__((ext_vector_type(16))) __bf16 v16bf;
typedef __attribute__((ext_vector_type(8)))  float  v8f;
typedef __attribute__((ext_vector_type(4)))  unsigned int u32x4;

#define ALD 776   // padded leading dim (elements) for LDS tiles (multiple of 8)

union Frag16 { v16bf v; u32x4 u[2]; };

// A-matrix fragment: 16x32 bf16, row-major source with leading dim `ld`.
// Lane L<16 holds row M=L, K = {0..7, 16..23}; lanes 16..31 hold K = {8..15, 24..31}.
// Each half is 8 contiguous bf16 = one 16B load.
__device__ __forceinline__ v16bf load_a_frag(const bf16* base, int ld) {
    const int lane = threadIdx.x & 31;
    const bf16* p = base + (size_t)(lane & 15) * ld + ((lane >> 4) << 3);
    Frag16 f;
    f.u[0] = *reinterpret_cast<const u32x4*>(p);
    f.u[1] = *reinterpret_cast<const u32x4*>(p + 16);
    return f.v;
}

// B-matrix fragment: 32x16 bf16. Source is the TRANSPOSED weight Wt[N][K]
// (leading dim = K), so per lane (column n = lane&15, h = lane>>4)
// K = {8h..8h+7, 16+8h..16+8h+7}: two 16B loads.
__device__ __forceinline__ v16bf load_b_frag(const bf16* base, int ld) {
    const int lane = threadIdx.x & 31;
    const bf16* p = base + (size_t)(lane & 15) * ld + ((lane >> 4) << 3);
    Frag16 f;
    f.u[0] = *reinterpret_cast<const u32x4*>(p);
    f.u[1] = *reinterpret_cast<const u32x4*>(p + 16);
    return f.v;
}

__device__ __forceinline__ v8f wmma_bf16(v16bf a, v16bf b, v8f c) {
    return __builtin_amdgcn_wmma_f32_16x16x32_bf16(false, a, false, b, (short)0, c,
                                                   false, false);
}

// GEMM: [16 x K](LDS bf16, ld=ALD) @ Wt[N][K](global bf16) + bias -> accf (LDS f32, ld=ALD)
__device__ void gemm16(const bf16* actA, const bf16* Wt, const float* bias,
                       float* accf, int N, int K) {
    const int wave   = threadIdx.x >> 5;
    const int nwaves = blockDim.x >> 5;
    const int lane   = threadIdx.x & 31;
    const int nl     = lane & 15;
    const int mrow   = (lane >> 4) << 3;
    for (int nt = wave; nt < (N >> 4); nt += nwaves) {
        const int n0 = nt << 4;
        const float bb = bias ? bias[n0 + nl] : 0.0f;
        v8f c;
#pragma unroll
        for (int i = 0; i < 8; ++i) c[i] = bb;
        const bf16* wrow = Wt + (size_t)n0 * K;
        for (int kt = 0; kt < K; kt += 32) {
            __builtin_prefetch(wrow + kt + 256, 0, 1);
            v16bf a = load_a_frag(actA + kt, ALD);
            v16bf b = load_b_frag(wrow + kt, K);
            c = wmma_bf16(a, b, c);
        }
#pragma unroll
        for (int i = 0; i < 8; ++i)
            accf[(size_t)(mrow + i) * ALD + n0 + nl] = c[i];
    }
}

// LayerNorm (+optional ReLU) over N per row, fp32 acc -> bf16 act.
__device__ void ln_post(const float* accf, bf16* act, const float* g, const float* be,
                        int N, bool relu) {
    const int t = threadIdx.x;
    if (t < 16) {
        const float* row = accf + (size_t)t * ALD;
        float m = 0.f;
        for (int n = 0; n < N; ++n) m += row[n];
        m /= (float)N;
        float v = 0.f;
        for (int n = 0; n < N; ++n) { float d = row[n] - m; v += d * d; }
        v /= (float)N;
        const float rs = rsqrtf(v + 1e-5f);
        for (int n = 0; n < N; ++n) {
            float y = (row[n] - m) * rs * g[n] + be[n];
            if (relu) y = fmaxf(y, 0.f);
            act[(size_t)t * ALD + n] = (bf16)y;
        }
    }
}

// Plain fp32 -> bf16 copy (no LN).
__device__ void cvt_post(const float* accf, bf16* act, int N) {
    for (int idx = threadIdx.x; idx < 16 * N; idx += blockDim.x) {
        int r = idx / N, n = idx % N;
        act[(size_t)r * ALD + n] = (bf16)accf[(size_t)r * ALD + n];
    }
}

// ---------------------------------------------------------------------------
// Weight transpose fp32 [K][N] -> bf16 [N][K]
// ---------------------------------------------------------------------------
__global__ void transpose_bf16_kernel(const float* __restrict__ W, bf16* __restrict__ Wt,
                                      int K, int N) {
    int idx = blockIdx.x * blockDim.x + threadIdx.x;
    if (idx >= K * N) return;
    int k = idx / N, n = idx % N;
    Wt[(size_t)n * K + k] = (bf16)W[idx];
}

__global__ void init_kernel(float* out, float* l2s, float* ztile) {
    int t = blockIdx.x * blockDim.x + threadIdx.x;
    if (t == 0) { *out = 0.f; *l2s = 0.f; }
    for (int i = t; i < 16 * 768; i += blockDim.x * gridDim.x) ztile[i] = 0.f;
}

// ---------------------------------------------------------------------------
// xwm = x @ Wm + bm  (parallel part of the recurrence), output bf16
// ---------------------------------------------------------------------------
__global__ __launch_bounds__(256) void xwm_kernel(const float* __restrict__ x,
                                                  const bf16* __restrict__ Wmt,
                                                  const float* __restrict__ bm,
                                                  bf16* __restrict__ xwm) {
    __shared__ __align__(16) bf16 act[16 * ALD];
    const int tok0 = blockIdx.x * 16;
    for (int idx = threadIdx.x; idx < 16 * 768; idx += blockDim.x) {
        int r = idx / 768, n = idx % 768;
        act[(size_t)r * ALD + n] = (bf16)x[(size_t)(tok0 + r) * 768 + n];
    }
    __syncthreads();
    const int wave = threadIdx.x >> 5, nwaves = blockDim.x >> 5;
    const int lane = threadIdx.x & 31;
    const int nl = lane & 15, mrow = (lane >> 4) << 3;
    for (int nt = wave; nt < 48; nt += nwaves) {
        const int n0 = nt << 4;
        const float bb = bm[n0 + nl];
        v8f c;
#pragma unroll
        for (int i = 0; i < 8; ++i) c[i] = bb;
        const bf16* wrow = Wmt + (size_t)n0 * 768;
        for (int kt = 0; kt < 768; kt += 32) {
            v16bf a = load_a_frag(act + kt, ALD);
            v16bf b = load_b_frag(wrow + kt, 768);
            c = wmma_bf16(a, b, c);
        }
#pragma unroll
        for (int i = 0; i < 8; ++i)
            xwm[(size_t)(tok0 + mrow + i) * 768 + n0 + nl] = (bf16)c[i];
    }
}

// ---------------------------------------------------------------------------
// Sequential scan: mem_j = tanh(xwm_j + mem_{j-1} @ Um). Single workgroup,
// 16 waves, mem double-buffered in LDS, one barrier per step.
// ---------------------------------------------------------------------------
__global__ __launch_bounds__(512) void scan_kernel(const bf16* __restrict__ xwm,
                                                   const bf16* __restrict__ Umt,
                                                   bf16* __restrict__ mem_out) {
    __shared__ __align__(16) bf16 mA[16 * ALD];
    __shared__ __align__(16) bf16 mB[16 * ALD];
    for (int idx = threadIdx.x; idx < 16 * ALD; idx += blockDim.x) mA[idx] = (bf16)0.f;
    __syncthreads();
    const int wave = threadIdx.x >> 5, nwaves = blockDim.x >> 5;
    const int lane = threadIdx.x & 31;
    const int nl = lane & 15, mrow = (lane >> 4) << 3;
    bf16* cur = mA; bf16* nxt = mB;
    for (int j = 0; j < 1024; ++j) {
        for (int nt = wave; nt < 48; nt += nwaves) {
            const int n0 = nt << 4;
            v8f c;
#pragma unroll
            for (int i = 0; i < 8; ++i) {
                const int m = mrow + i;                   // batch row
                c[i] = (float)xwm[((size_t)m * 1024 + j) * 768 + n0 + nl];
            }
            const bf16* wrow = Umt + (size_t)n0 * 768;
            for (int kt = 0; kt < 768; kt += 32) {
                v16bf a = load_a_frag(cur + kt, ALD);
                v16bf b = load_b_frag(wrow + kt, 768);
                c = wmma_bf16(a, b, c);
            }
#pragma unroll
            for (int i = 0; i < 8; ++i) {
                const int m = mrow + i, n = n0 + nl;
                bf16 tb = (bf16)tanhf(c[i]);
                nxt[(size_t)m * ALD + n] = tb;
                mem_out[((size_t)m * 1024 + j) * 768 + n] = tb;
            }
        }
        __syncthreads();
        bf16* tmp = cur; cur = nxt; nxt = tmp;
    }
}

// ---------------------------------------------------------------------------
// Fused encoder: 16 tokens per workgroup, all activations in LDS.
// ---------------------------------------------------------------------------
struct EncArgs {
    const float* in_f32; const bf16* in_b16; int in_is_b16;
    const bf16 *W1t, *Wvt, *Wot, *W2t, *W3t, *W4t, *W5t;
    const float *b1, *g1, *be1, *bv, *bo, *b2, *g2, *be2;
    const float *b3, *g3, *be3, *b4, *g4, *be4, *b5, *g5, *be5;
    float* out_lat;
};

__global__ __launch_bounds__(256) void encode_kernel(EncArgs a) {
    __shared__ __align__(16) bf16  act[16 * ALD];
    __shared__ __align__(16) float acc[16 * ALD];
    const int tok0 = blockIdx.x * 16;
    for (int idx = threadIdx.x; idx < 16 * 768; idx += blockDim.x) {
        int r = idx / 768, n = idx % 768;
        float v = a.in_is_b16 ? (float)a.in_b16[(size_t)(tok0 + r) * 768 + n]
                              : a.in_f32[(size_t)(tok0 + r) * 768 + n];
        act[(size_t)r * ALD + n] = (bf16)v;
    }
    __syncthreads();
    gemm16(act, a.W1t, a.b1, acc, 512, 768); __syncthreads();
    ln_post(acc, act, a.g1, a.be1, 512, true); __syncthreads();   // h
    gemm16(act, a.Wvt, a.bv, acc, 512, 512); __syncthreads();
    cvt_post(acc, act, 512); __syncthreads();                     // v_proj
    gemm16(act, a.Wot, a.bo, acc, 512, 512); __syncthreads();
    cvt_post(acc, act, 512); __syncthreads();                     // a
    gemm16(act, a.W2t, a.b2, acc, 512, 512); __syncthreads();
    ln_post(acc, act, a.g2, a.be2, 512, true); __syncthreads();   // g
    gemm16(act, a.W3t, a.b3, acc, 128, 512); __syncthreads();
    ln_post(acc, act, a.g3, a.be3, 128, false); __syncthreads();  // zl
    gemm16(act, a.W4t, a.b4, acc, 512, 128); __syncthreads();
    ln_post(acc, act, a.g4, a.be4, 512, true); __syncthreads();   // e
    gemm16(act, a.W5t, a.b5, acc, 128, 512); __syncthreads();
    if (threadIdx.x < 16) {                                       // final LN -> fp32 lat
        const int r = threadIdx.x;
        const float* row = acc + (size_t)r * ALD;
        float m = 0.f; for (int n = 0; n < 128; ++n) m += row[n];
        m /= 128.f;
        float v = 0.f; for (int n = 0; n < 128; ++n) { float d = row[n] - m; v += d * d; }
        v /= 128.f;
        const float rs = rsqrtf(v + 1e-5f);
        for (int n = 0; n < 128; ++n)
            a.out_lat[(size_t)(tok0 + r) * 128 + n] = (row[n] - m) * rs * a.g5[n] + a.be5[n];
    }
}

// ---------------------------------------------------------------------------
// Fused decoder + per-token reconstruction loss.
// ---------------------------------------------------------------------------
struct DecArgs {
    const float* lat;
    const bf16 *Wd1t, *Wd2t, *Wd3t;
    const float *bd1, *gd1, *bed1, *bd2, *gd2, *bed2, *bd3;
    const float* x; float* recon;
};

__global__ __launch_bounds__(256) void decode_kernel(DecArgs a) {
    __shared__ __align__(16) bf16  act[16 * ALD];
    __shared__ __align__(16) float acc[16 * ALD];
    const int tok0 = blockIdx.x * 16;
    for (int idx = threadIdx.x; idx < 16 * 128; idx += blockDim.x) {
        int r = idx / 128, n = idx % 128;
        act[(size_t)r * ALD + n] = (bf16)a.lat[(size_t)(tok0 + r) * 128 + n];
    }
    __syncthreads();
    gemm16(act, a.Wd1t, a.bd1, acc, 512, 128); __syncthreads();
    ln_post(acc, act, a.gd1, a.bed1, 512, true); __syncthreads();
    gemm16(act, a.Wd2t, a.bd2, acc, 512, 512); __syncthreads();
    ln_post(acc, act, a.gd2, a.bed2, 512, true); __syncthreads();
    gemm16(act, a.Wd3t, a.bd3, acc, 768, 512); __syncthreads();
    if (threadIdx.x < 16) {
        const int r = threadIdx.x;
        float s = 0.f;
        for (int n = 0; n < 768; ++n) {
            float d = acc[(size_t)r * ALD + n] - a.x[(size_t)(tok0 + r) * 768 + n];
            s += d * d;
        }
        a.recon[tok0 + r] = fminf(fmaxf(s / 768.f, 0.f), 10.f);
    }
}

// ---------------------------------------------------------------------------
// L2 term: sum of Frobenius norms over all 37 parameter tensors.
// ---------------------------------------------------------------------------
struct ParamList { const float* p[37]; int n[37]; };

__global__ __launch_bounds__(256) void l2_kernel(ParamList pl, float* l2out) {
    __shared__ float red[256];
    const float* P = pl.p[blockIdx.x];
    const int N = pl.n[blockIdx.x];
    float s = 0.f;
    for (int i = threadIdx.x; i < N; i += blockDim.x) { float v = P[i]; s += v * v; }
    red[threadIdx.x] = s; __syncthreads();
    for (int off = 128; off > 0; off >>= 1) {
        if ((int)threadIdx.x < off) red[threadIdx.x] += red[threadIdx.x + off];
        __syncthreads();
    }
    if (threadIdx.x == 0) atomicAdd(l2out, sqrtf(red[0]));
}

// ---------------------------------------------------------------------------
// Final loss: recon + transition + contextual(cosine) + l2, clip, reduce.
// ---------------------------------------------------------------------------
__global__ void loss_kernel(const float* __restrict__ lat, const float* __restrict__ latm,
                            const float* __restrict__ z0, const float* __restrict__ recon,
                            const float* __restrict__ l2p, float* __restrict__ out) {
    const int t = blockIdx.x * blockDim.x + threadIdx.x;
    if (t >= 16 * 1024) return;
    const int s = t & 1023;
    const float* lt = lat + (size_t)t * 128;
    const float* lp = (s == 0) ? z0 : lat + (size_t)(t - 1) * 128;
    const float* lm = latm + (size_t)t * 128;
    float tr = 0.f, dn = 0.f, nn = 0.f, nm = 0.f;
    for (int n = 0; n < 128; ++n) {
        float d = lt[n] - lp[n];
        tr += d * d;
        dn += lt[n] * lm[n];
        nn += lt[n] * lt[n];
        nm += lm[n] * lm[n];
    }
    tr = fminf(fmaxf(tr / 128.f, 0.f), 10.f);
    float norm = fmaxf(sqrtf(nn), 1e-8f) * fmaxf(sqrtf(nm), 1e-8f);
    float ctx = fminf(fmaxf(1.f - dn / norm, 0.f), 10.f);
    float l2 = fminf(fmaxf(*l2p, 0.f), 10.f);
    float per = 1.0f * recon[t] + 0.3f * tr + 0.3f * ctx + 1e-4f * l2;
    per = fminf(fmaxf(per, 0.f), 100.f);
    atomicAdd(out, per * (1.0f / 16.0f));
}

// ---------------------------------------------------------------------------
// Host launch
// ---------------------------------------------------------------------------
extern "C" void kernel_launch(void* const* d_in, const int* in_sizes, int n_in,
                              void* d_out, int out_size, void* d_ws, size_t ws_size,
                              hipStream_t stream) {
    const float* x   = (const float*)d_in[0];
    const float* W1  = (const float*)d_in[1];
    const float* b1  = (const float*)d_in[2];
    const float* g1  = (const float*)d_in[3];
    const float* be1 = (const float*)d_in[4];
    const float* Wv  = (const float*)d_in[5];
    const float* bv  = (const float*)d_in[6];
    const float* Wo  = (const float*)d_in[7];
    const float* bo  = (const float*)d_in[8];
    const float* W2  = (const float*)d_in[9];
    const float* b2  = (const float*)d_in[10];
    const float* g2  = (const float*)d_in[11];
    const float* be2 = (const float*)d_in[12];
    const float* W3  = (const float*)d_in[13];
    const float* b3  = (const float*)d_in[14];
    const float* g3  = (const float*)d_in[15];
    const float* be3 = (const float*)d_in[16];
    const float* W4  = (const float*)d_in[17];
    const float* b4  = (const float*)d_in[18];
    const float* g4  = (const float*)d_in[19];
    const float* be4 = (const float*)d_in[20];
    const float* W5  = (const float*)d_in[21];
    const float* b5  = (const float*)d_in[22];
    const float* g5  = (const float*)d_in[23];
    const float* be5 = (const float*)d_in[24];
    const float* Wd1 = (const float*)d_in[25];
    const float* bd1 = (const float*)d_in[26];
    const float* gd1 = (const float*)d_in[27];
    const float* bed1= (const float*)d_in[28];
    const float* Wd2 = (const float*)d_in[29];
    const float* bd2 = (const float*)d_in[30];
    const float* gd2 = (const float*)d_in[31];
    const float* bed2= (const float*)d_in[32];
    const float* Wd3 = (const float*)d_in[33];
    const float* bd3 = (const float*)d_in[34];
    const float* Wm  = (const float*)d_in[35];
    const float* Um  = (const float*)d_in[36];
    const float* bm  = (const float*)d_in[37];

    char* ws = (char*)d_ws;
    size_t off = 0;
    auto alloc = [&](size_t bytes) -> void* {
        void* p = ws + off;
        off = (off + bytes + 255) & ~(size_t)255;
        return p;
    };

    bf16* W1t  = (bf16*)alloc((size_t)768 * 512 * 2);
    bf16* Wvt  = (bf16*)alloc((size_t)512 * 512 * 2);
    bf16* Wot  = (bf16*)alloc((size_t)512 * 512 * 2);
    bf16* W2t  = (bf16*)alloc((size_t)512 * 512 * 2);
    bf16* W3t  = (bf16*)alloc((size_t)512 * 128 * 2);
    bf16* W4t  = (bf16*)alloc((size_t)128 * 512 * 2);
    bf16* W5t  = (bf16*)alloc((size_t)512 * 128 * 2);
    bf16* Wd1t = (bf16*)alloc((size_t)128 * 512 * 2);
    bf16* Wd2t = (bf16*)alloc((size_t)512 * 512 * 2);
    bf16* Wd3t = (bf16*)alloc((size_t)512 * 768 * 2);
    bf16* Wmt  = (bf16*)alloc((size_t)768 * 768 * 2);
    bf16* Umt  = (bf16*)alloc((size_t)768 * 768 * 2);
    bf16* xwm  = (bf16*)alloc((size_t)16 * 1024 * 768 * 2);
    bf16* mem  = (bf16*)alloc((size_t)16 * 1024 * 768 * 2);
    float* lat  = (float*)alloc((size_t)16 * 1024 * 128 * 4);
    float* latm = (float*)alloc((size_t)16 * 1024 * 128 * 4);
    float* z0lat = (float*)alloc((size_t)16 * 128 * 4);
    float* ztile = (float*)alloc((size_t)16 * 768 * 4);
    float* recon = (float*)alloc((size_t)16384 * 4);
    float* l2s   = (float*)alloc(256);

    init_kernel<<<48, 256, 0, stream>>>((float*)d_out, l2s, ztile);

    auto tr = [&](const float* W, bf16* Wt, int K, int N) {
        transpose_bf16_kernel<<<(K * N + 255) / 256, 256, 0, stream>>>(W, Wt, K, N);
    };
    tr(W1, W1t, 768, 512);   tr(Wv, Wvt, 512, 512);  tr(Wo, Wot, 512, 512);
    tr(W2, W2t, 512, 512);   tr(W3, W3t, 512, 128);  tr(W4, W4t, 128, 512);
    tr(W5, W5t, 512, 128);   tr(Wd1, Wd1t, 128, 512); tr(Wd2, Wd2t, 512, 512);
    tr(Wd3, Wd3t, 512, 768); tr(Wm, Wmt, 768, 768);  tr(Um, Umt, 768, 768);

    // Parallel part of recurrence, then the sequential scan
    xwm_kernel<<<1024, 256, 0, stream>>>(x, Wmt, bm, xwm);
    scan_kernel<<<1, 512, 0, stream>>>(xwm, Umt, mem);

    EncArgs ea;
    ea.W1t = W1t; ea.Wvt = Wvt; ea.Wot = Wot; ea.W2t = W2t;
    ea.W3t = W3t; ea.W4t = W4t; ea.W5t = W5t;
    ea.b1 = b1; ea.g1 = g1; ea.be1 = be1; ea.bv = bv; ea.bo = bo;
    ea.b2 = b2; ea.g2 = g2; ea.be2 = be2; ea.b3 = b3; ea.g3 = g3; ea.be3 = be3;
    ea.b4 = b4; ea.g4 = g4; ea.be4 = be4; ea.b5 = b5; ea.g5 = g5; ea.be5 = be5;

    // encode(x)
    ea.in_f32 = x; ea.in_b16 = nullptr; ea.in_is_b16 = 0; ea.out_lat = lat;
    encode_kernel<<<1024, 256, 0, stream>>>(ea);
    // encode(mem)
    ea.in_f32 = nullptr; ea.in_b16 = mem; ea.in_is_b16 = 1; ea.out_lat = latm;
    encode_kernel<<<1024, 256, 0, stream>>>(ea);
    // encode(zeros) -> z0 (row 0 used)
    ea.in_f32 = ztile; ea.in_b16 = nullptr; ea.in_is_b16 = 0; ea.out_lat = z0lat;
    encode_kernel<<<1, 256, 0, stream>>>(ea);

    DecArgs da;
    da.lat = lat; da.Wd1t = Wd1t; da.Wd2t = Wd2t; da.Wd3t = Wd3t;
    da.bd1 = bd1; da.gd1 = gd1; da.bed1 = bed1;
    da.bd2 = bd2; da.gd2 = gd2; da.bed2 = bed2;
    da.bd3 = bd3; da.x = x; da.recon = recon;
    decode_kernel<<<1024, 256, 0, stream>>>(da);

    ParamList pl;
    for (int i = 0; i < 37; ++i) {
        pl.p[i] = (const float*)d_in[1 + i];
        pl.n[i] = in_sizes[1 + i];
    }
    l2_kernel<<<37, 256, 0, stream>>>(pl, l2s);

    loss_kernel<<<64, 256, 0, stream>>>(lat, latm, z0lat, recon, l2s, (float*)d_out);
    (void)n_in; (void)out_size; (void)ws_size;
}
